// FusionTransformer_62457414418914
// MI455X (gfx1250) — matrix operand order repaired and checked by reading, not compiled, not run on hardware
//
#include <hip/hip_runtime.h>

// ---------------------------------------------------------------------------
// 2-layer transformer encoder forward for MI455X (gfx1250, wave32, WMMA).
// GEMMs: v_wmma_f32_16x16x32_bf16, operands DMA'd into double-buffered LDS by
// the Tensor Data Mover (tensor_load_to_lds + s_wait_tensorcnt).
// Attention: flash-style online softmax (never materializes the 1.07 GB
// score tensor).
// ---------------------------------------------------------------------------

#define DM   1024
#define DFF  4096
#define NH   16
#define DH   64
#define SEQL 2048
#define NB   4
#define NTOK (NB * SEQL)   // 8192 tokens

typedef __attribute__((ext_vector_type(16))) __bf16        v16bf;
typedef __attribute__((ext_vector_type(8)))  float         v8f;
typedef __attribute__((ext_vector_type(4)))  unsigned int  v4u;
typedef __attribute__((ext_vector_type(8)))  int           v8i;
typedef __attribute__((ext_vector_type(4)))  int           v4i;

__device__ __forceinline__ unsigned short f2bf(float f) {
  unsigned int u = __float_as_uint(f);
  u += 0x7fffu + ((u >> 16) & 1u);          // round-to-nearest-even
  return (unsigned short)(u >> 16);
}

__device__ __forceinline__ v8f wmma_bf16(v16bf a, v16bf b, v8f c) {
  // (neg_a, A, neg_b, B, c_mod, C, reuse_a, reuse_b)
  return __builtin_amdgcn_wmma_f32_16x16x32_bf16(false, a, false, b, (short)0, c,
                                                 false, false);
}

// ---------------------------------------------------------------------------
// TDM: DMA a 2D bf16 tile [tile_rows x tile_k] (row stride = row_len elems)
// from global into LDS at byte offset lds_byte_addr (packed row-major,
// tile_k elems per row). Descriptor packing per CDNA5 ISA
// D# group0/group1; groups 2/3 + extra group zero (2D, tile_dim2=0 unused).
// This toolchain's builtin is the 6-arg clang-23 form.
// ---------------------------------------------------------------------------
__device__ __forceinline__ void tdm_load_2d(unsigned lds_byte_addr,
                                            const unsigned short* g,
                                            unsigned row_len, unsigned rows_total,
                                            unsigned tile_k, unsigned tile_rows) {
  unsigned long long ga = (unsigned long long)g;
  v4u g0;
  g0.x = 1u;                                  // count = 1 valid descriptor
  g0.y = lds_byte_addr;                       // lds_addr (bytes)
  g0.z = (unsigned)ga;                        // global_addr[31:0]
  g0.w = (unsigned)(ga >> 32) | (2u << 30);   // global_addr[56:32] | type=2
  v8i g1;
  g1[0] = (int)(1u << 16);                    // data_size = 1 -> 2 bytes
  g1[1] = (int)((row_len & 0xFFFFu) << 16);   // tensor_dim0[15:0]
  g1[2] = (int)((row_len >> 16) | ((rows_total & 0xFFFFu) << 16)); // td0 hi | td1 lo
  g1[3] = (int)((rows_total >> 16) | (tile_k << 16));              // td1 hi | tile_dim0
  g1[4] = (int)tile_rows;                     // tile_dim1 (tile_dim2 = 0)
  g1[5] = (int)row_len;                       // tensor_dim0_stride[31:0]
  g1[6] = 0;                                  // stride0 hi | stride1 lo
  g1[7] = 0;
  v4i z4 = {0, 0, 0, 0};
  v8i z8 = {0, 0, 0, 0, 0, 0, 0, 0};
  __builtin_amdgcn_tensor_load_to_lds(g0, g1, z4, z4, z8, 0);
}

// ---------------------------------------------------------------------------
// Weight prep: W[din][dout] fp32 -> Wt[dout][din] bf16 (transposed so TDM
// tiles and GEMM B-fragments are contiguous along K). Coalesced writes.
// ---------------------------------------------------------------------------
__global__ __launch_bounds__(256)
void conv_transpose(const float* __restrict__ W, unsigned short* __restrict__ Wt,
                    int din, int dout) {
  const size_t idx = (size_t)blockIdx.x * 256 + threadIdx.x;
  const int n = (int)(idx / din);
  const int k = (int)(idx % din);
  Wt[idx] = f2bf(W[(size_t)k * dout + n]);
}

// ---------------------------------------------------------------------------
// LayerNorm over D=1024, one block per token.
// ---------------------------------------------------------------------------
template <bool BF16OUT>
__global__ __launch_bounds__(256)
void layernorm_k(const float* __restrict__ x, const float* __restrict__ gam,
                 const float* __restrict__ bet, void* __restrict__ out) {
  __shared__ float red[8];
  const int t = threadIdx.x;
  const size_t base = (size_t)blockIdx.x * DM;
  float v[4];
  float s = 0.0f;
#pragma unroll
  for (int i = 0; i < 4; ++i) { v[i] = x[base + t + 256 * i]; s += v[i]; }
#pragma unroll
  for (int d = 16; d >= 1; d >>= 1) s += __shfl_xor(s, d, 32);
  if ((t & 31) == 0) red[t >> 5] = s;
  __syncthreads();
  float tot = 0.0f;
#pragma unroll
  for (int i = 0; i < 8; ++i) tot += red[i];
  const float mu = tot * (1.0f / DM);
  float s2 = 0.0f;
#pragma unroll
  for (int i = 0; i < 4; ++i) { float dd = v[i] - mu; s2 += dd * dd; }
#pragma unroll
  for (int d = 16; d >= 1; d >>= 1) s2 += __shfl_xor(s2, d, 32);
  __syncthreads();
  if ((t & 31) == 0) red[t >> 5] = s2;
  __syncthreads();
  float vt = 0.0f;
#pragma unroll
  for (int i = 0; i < 8; ++i) vt += red[i];
  const float inv = rsqrtf(vt * (1.0f / DM) + 1e-5f);
#pragma unroll
  for (int i = 0; i < 4; ++i) {
    const int c = t + 256 * i;
    const float y = (v[i] - mu) * inv * gam[c] + bet[c];
    if (BF16OUT) ((unsigned short*)out)[base + c] = f2bf(y);
    else         ((float*)out)[base + c] = y;
  }
}

// ---------------------------------------------------------------------------
// GEMM: out[M][N] = A[M][K](bf16) * Wt[N][K](bf16, pre-transposed) + bias,
// optional fp32 residual add (in-place safe) and exact-erf GELU.
//
// Block: 256 thr = 8 waves (4 M-groups x 2 N-groups), tile 128x128, K-step 64.
// Each wave: 32 rows x 64 cols = eight 16x16 fp32 accumulators, 16 WMMAs per
// K-step. Operand tiles (A 128x64, B 128x64) are DMA'd by the TDM into
// double-buffered dynamic LDS (64 KB; no static LDS -> descriptor offsets
// are exact: A at 0/16K, B at 32K/48K). Wave 0 issues/waits (TENSORcnt),
// barriers publish to the block.
//
// Fragment layouts per CDNA5 ISA (wave32):
//   A 16x32: lane row=l&15, h=l>>4 holds K = 8h+{0..7}, 16+8h+{0..7}
//   B 32x16: lane col=l&15, h=l>>4 holds K = 8h+{0..7}, 16+8h+{0..7}
//   C 16x16: vgpr i -> row i+8h, col = l&15
// ---------------------------------------------------------------------------
template <bool GELU, bool BF16OUT, bool RES>
__global__ __launch_bounds__(256)
void gemm_bf16(const unsigned short* __restrict__ A,
               const unsigned short* __restrict__ Wt,
               const float* __restrict__ bias,
               const float* res, void* out, int M, int N, int K) {
  extern __shared__ __align__(16) unsigned short smem[];

  const int nb    = N >> 7;
  const int mbase = (blockIdx.x / nb) * 128;
  const int nbase = (blockIdx.x % nb) * 128;
  const int w    = threadIdx.x >> 5;
  const int lane = threadIdx.x & 31;
  const int wm = w >> 1;            // 0..3 : 32-row group
  const int wn = w & 1;             // 0..1 : 64-col group
  const int r  = lane & 15;
  const int hh = lane >> 4;

  const unsigned short* Ag = A  + (size_t)mbase * K;
  const unsigned short* Bg = Wt + (size_t)nbase * K;

  v8f cc[2][4] = {};

  const bool issuer = (threadIdx.x < 32);   // wave 0 drives the TDM
  if (issuer) {
    tdm_load_2d(0u,      Ag, (unsigned)K, 128u, 64u, 128u);
    tdm_load_2d(32768u,  Bg, (unsigned)K, 128u, 64u, 128u);
  }

  const int steps = K >> 6;
  for (int s = 0; s < steps; ++s) {
    const int cur = s & 1;
    if (issuer) {
      if (s + 1 < steps) {
        const unsigned nxtOff = cur ? 0u : 16384u;     // byte offset of spare buf
        tdm_load_2d(nxtOff,           Ag + (s + 1) * 64, (unsigned)K, 128u, 64u, 128u);
        tdm_load_2d(32768u + nxtOff,  Bg + (s + 1) * 64, (unsigned)K, 128u, 64u, 128u);
        __builtin_amdgcn_s_wait_tensorcnt((short)2);   // tile s complete
      } else {
        __builtin_amdgcn_s_wait_tensorcnt((short)0);
      }
    }
    __syncthreads();                                   // publish tile s

    const unsigned short* As = smem + (cur ? 8192 : 0);           // elems
    const unsigned short* Bs = smem + 16384 + (cur ? 8192 : 0);   // elems
#pragma unroll
    for (int c = 0; c < 2; ++c) {                      // two 32-wide K chunks
      v16bf a0, a1;
      const unsigned short* ar0 = As + (wm * 32 + r) * 64 + c * 32;
      const unsigned short* ar1 = ar0 + 16 * 64;
      ((uint4*)&a0)[0] = *(const uint4*)(ar0 + 8 * hh);
      ((uint4*)&a0)[1] = *(const uint4*)(ar0 + 16 + 8 * hh);
      ((uint4*)&a1)[0] = *(const uint4*)(ar1 + 8 * hh);
      ((uint4*)&a1)[1] = *(const uint4*)(ar1 + 16 + 8 * hh);
#pragma unroll
      for (int j = 0; j < 4; ++j) {
        v16bf b;
        const unsigned short* br =
            Bs + (wn * 64 + j * 16 + r) * 64 + c * 32 + 8 * hh;
        ((uint4*)&b)[0] = *(const uint4*)br;
        ((uint4*)&b)[1] = *(const uint4*)(br + 16);
        cc[0][j] = wmma_bf16(a0, b, cc[0][j]);
        cc[1][j] = wmma_bf16(a1, b, cc[1][j]);
      }
    }
    __syncthreads();          // all reads done before next DMA overwrites buf
  }

  float bj[4];
#pragma unroll
  for (int j = 0; j < 4; ++j) bj[j] = bias[nbase + wn * 64 + j * 16 + r];

#pragma unroll
  for (int mi = 0; mi < 2; ++mi) {
#pragma unroll
    for (int i = 0; i < 8; ++i) {
      const int row = mbase + wm * 32 + mi * 16 + i + 8 * hh;
#pragma unroll
      for (int j = 0; j < 4; ++j) {
        const int col = nbase + wn * 64 + j * 16 + r;
        float acc = cc[mi][j][i] + bj[j];
        if (RES)  acc += res[(size_t)row * N + col];
        if (GELU) acc = 0.5f * acc * (1.0f + erff(acc * 0.70710678118654752f));
        if (BF16OUT) ((unsigned short*)out)[(size_t)row * N + col] = f2bf(acc);
        else         ((float*)out)[(size_t)row * N + col]          = acc;
      }
    }
  }
  (void)M;
}

// ---------------------------------------------------------------------------
// Flash attention: block = 128 thr (4 waves), one (b,h) + 64 query rows per
// block; each wave owns a 16-row query tile. Streams keys in steps of 32:
//   S(16x32) = Q(16x64) Kt -> 4 WMMAs; online softmax via shfl_xor row groups;
//   O(16x64) += P(16x32) V -> 4 WMMAs (P re-striped C->A layout through LDS,
//   V staged transposed so B-fragments are contiguous).
// ---------------------------------------------------------------------------
__global__ __launch_bounds__(128)
void attn_flash(const unsigned short* __restrict__ qg,
                const unsigned short* __restrict__ kg,
                const unsigned short* __restrict__ vg,
                unsigned short* __restrict__ og) {
  __shared__ __align__(16) unsigned short Ksm[32 * 64];   // [key][d]
  __shared__ __align__(16) unsigned short Vtm[64 * 32];   // [d][key]
  __shared__ __align__(16) unsigned short Psm[4 * 16 * 32];

  const int bh  = blockIdx.x >> 5;          // 64 = B*H
  const int qg4 = blockIdx.x & 31;          // SEQL/64 groups
  const int b = bh >> 4;
  const int h = bh & 15;
  const int w    = threadIdx.x >> 5;
  const int lane = threadIdx.x & 31;
  const int r  = lane & 15;
  const int hh = lane >> 4;
  const int qbase = qg4 * 64 + w * 16;

  v16bf aq0, aq1;
  {
    const unsigned short* qr = qg + ((size_t)(b * SEQL + qbase + r)) * DM + h * DH;
    ((uint4*)&aq0)[0] = *(const uint4*)(qr + 16 * hh);
    ((uint4*)&aq0)[1] = *(const uint4*)(qr + 8 + 16 * hh);
    ((uint4*)&aq1)[0] = *(const uint4*)(qr + 32 + 16 * hh);
    ((uint4*)&aq1)[1] = *(const uint4*)(qr + 40 + 16 * hh);
  }

  float mrow[8], lrow[8];
  v8f oc[4] = {};
#pragma unroll
  for (int i = 0; i < 8; ++i) { mrow[i] = -3.0e38f; lrow[i] = 0.0f; }

  const int srow = threadIdx.x >> 2, sseg = threadIdx.x & 3;

  for (int kb = 0; kb < SEQL; kb += 32) {
    {  // cooperative staging: K row-major, V transposed
      const unsigned short* ks =
          kg + ((size_t)(b * SEQL + kb + srow)) * DM + h * DH + sseg * 16;
      *(uint4*)(Ksm + srow * 64 + sseg * 16)     = ((const uint4*)ks)[0];
      *(uint4*)(Ksm + srow * 64 + sseg * 16 + 8) = ((const uint4*)ks)[1];
      const unsigned short* vs =
          vg + ((size_t)(b * SEQL + kb + srow)) * DM + h * DH + sseg * 16;
#pragma unroll
      for (int e = 0; e < 16; ++e)
        Vtm[(sseg * 16 + e) * 32 + srow] = vs[e];
    }
    __syncthreads();

    v8f s0 = {}, s1 = {};
    {
      v16bf bk;
      const unsigned short* k0 = Ksm + r * 64 + 16 * hh;
      ((uint4*)&bk)[0] = *(const uint4*)(k0);
      ((uint4*)&bk)[1] = *(const uint4*)(k0 + 8);
      s0 = wmma_bf16(aq0, bk, s0);
      ((uint4*)&bk)[0] = *(const uint4*)(k0 + 32);
      ((uint4*)&bk)[1] = *(const uint4*)(k0 + 40);
      s0 = wmma_bf16(aq1, bk, s0);
      const unsigned short* k1 = Ksm + (16 + r) * 64 + 16 * hh;
      ((uint4*)&bk)[0] = *(const uint4*)(k1);
      ((uint4*)&bk)[1] = *(const uint4*)(k1 + 8);
      s1 = wmma_bf16(aq0, bk, s1);
      ((uint4*)&bk)[0] = *(const uint4*)(k1 + 32);
      ((uint4*)&bk)[1] = *(const uint4*)(k1 + 40);
      s1 = wmma_bf16(aq1, bk, s1);
    }

#pragma unroll
    for (int i = 0; i < 8; ++i) {
      float a0 = s0[i] * 0.125f;      // 1/sqrt(64)
      float a1 = s1[i] * 0.125f;
      float mx = fmaxf(a0, a1);
#pragma unroll
      for (int d = 8; d >= 1; d >>= 1) mx = fmaxf(mx, __shfl_xor(mx, d, 32));
      const float mn    = fmaxf(mrow[i], mx);
      const float alpha = __expf(mrow[i] - mn);
      const float p0 = __expf(a0 - mn);
      const float p1 = __expf(a1 - mn);
      float rs = p0 + p1;
#pragma unroll
      for (int d = 8; d >= 1; d >>= 1) rs += __shfl_xor(rs, d, 32);
      lrow[i] = lrow[i] * alpha + rs;
      mrow[i] = mn;
#pragma unroll
      for (int j = 0; j < 4; ++j) oc[j][i] *= alpha;
      Psm[(w * 16 + i + 8 * hh) * 32 + r]      = f2bf(p0);
      Psm[(w * 16 + i + 8 * hh) * 32 + 16 + r] = f2bf(p1);
    }
    __syncthreads();

    v16bf pa;
    const unsigned short* pr = Psm + (w * 16 + r) * 32;
    ((uint4*)&pa)[0] = *(const uint4*)(pr + 16 * hh);
    ((uint4*)&pa)[1] = *(const uint4*)(pr + 8 + 16 * hh);

#pragma unroll
    for (int j = 0; j < 4; ++j) {
      v16bf bv;
      const unsigned short* vr = Vtm + (j * 16 + r) * 32 + 16 * hh;
      ((uint4*)&bv)[0] = *(const uint4*)vr;
      ((uint4*)&bv)[1] = *(const uint4*)(vr + 8);
      oc[j] = wmma_bf16(pa, bv, oc[j]);
    }
    __syncthreads();
  }

#pragma unroll
  for (int i = 0; i < 8; ++i) {
    const float invl = 1.0f / lrow[i];
    const size_t base = ((size_t)(b * SEQL + qbase + i + 8 * hh)) * DM + h * DH;
#pragma unroll
    for (int j = 0; j < 4; ++j)
      og[base + j * 16 + r] = f2bf(oc[j][i] * invl);
  }
}

// ---------------------------------------------------------------------------
// Orchestration. Input order (jax pytree, sorted dict keys):
//   [0]=x ; per layer (16): W1,W2,Wk,Wo,Wq,Wv,b1,b2,bk,bo,bq,bv,n1b,n1g,n2b,n2g
//   [33]=norm_b [34]=norm_g
// ---------------------------------------------------------------------------
extern "C" void kernel_launch(void* const* d_in, const int* in_sizes, int n_in,
                              void* d_out, int out_size, void* d_ws, size_t ws_size,
                              hipStream_t stream) {
  (void)in_sizes; (void)n_in; (void)out_size; (void)ws_size;

  char* p = (char*)d_ws;
  float*          xw  = (float*)p;          p += (size_t)NTOK * DM * 4;   // residual
  unsigned short* lnb = (unsigned short*)p; p += (size_t)NTOK * DM * 2;   // LN / attn out
  unsigned short* qb  = (unsigned short*)p; p += (size_t)NTOK * DM * 2;
  unsigned short* kb  = (unsigned short*)p; p += (size_t)NTOK * DM * 2;
  unsigned short* vb  = (unsigned short*)p; p += (size_t)NTOK * DM * 2;
  unsigned short* hb  = (unsigned short*)p; p += (size_t)NTOK * DFF * 2;  // FFN hidden
  unsigned short* wqt = (unsigned short*)p; p += (size_t)DM * DM * 2;
  unsigned short* wkt = (unsigned short*)p; p += (size_t)DM * DM * 2;
  unsigned short* wvt = (unsigned short*)p; p += (size_t)DM * DM * 2;
  unsigned short* wot = (unsigned short*)p; p += (size_t)DM * DM * 2;
  unsigned short* w1t = (unsigned short*)p; p += (size_t)DM * DFF * 2;
  unsigned short* w2t = (unsigned short*)p; p += (size_t)DFF * DM * 2;

  (void)hipMemcpyAsync(xw, d_in[0], (size_t)NTOK * DM * 4,
                       hipMemcpyDeviceToDevice, stream);

  const int    gemmDM  = (NTOK / 128) * (DM / 128);    // 512 blocks
  const int    gemmFF  = (NTOK / 128) * (DFF / 128);   // 2048 blocks
  const size_t gemmLds = 65536;                        // 2x(A 16K + B 16K)

  for (int layer = 0; layer < 2; ++layer) {
    const int base = 1 + layer * 16;
    const float* W1  = (const float*)d_in[base + 0];
    const float* W2  = (const float*)d_in[base + 1];
    const float* Wk  = (const float*)d_in[base + 2];
    const float* Wo  = (const float*)d_in[base + 3];
    const float* Wq  = (const float*)d_in[base + 4];
    const float* Wv  = (const float*)d_in[base + 5];
    const float* b1  = (const float*)d_in[base + 6];
    const float* b2  = (const float*)d_in[base + 7];
    const float* bkp = (const float*)d_in[base + 8];
    const float* bo  = (const float*)d_in[base + 9];
    const float* bq  = (const float*)d_in[base + 10];
    const float* bvp = (const float*)d_in[base + 11];
    const float* n1b = (const float*)d_in[base + 12];
    const float* n1g = (const float*)d_in[base + 13];
    const float* n2b = (const float*)d_in[base + 14];
    const float* n2g = (const float*)d_in[base + 15];

    conv_transpose<<<DM * DM / 256, 256, 0, stream>>>(Wq, wqt, DM, DM);
    conv_transpose<<<DM * DM / 256, 256, 0, stream>>>(Wk, wkt, DM, DM);
    conv_transpose<<<DM * DM / 256, 256, 0, stream>>>(Wv, wvt, DM, DM);
    conv_transpose<<<DM * DM / 256, 256, 0, stream>>>(Wo, wot, DM, DM);
    conv_transpose<<<DM * DFF / 256, 256, 0, stream>>>(W1, w1t, DM, DFF);
    conv_transpose<<<DFF * DM / 256, 256, 0, stream>>>(W2, w2t, DFF, DM);

    layernorm_k<true><<<NTOK, 256, 0, stream>>>(xw, n1g, n1b, lnb);

    gemm_bf16<false, true, false><<<gemmDM, 256, gemmLds, stream>>>(lnb, wqt, bq,  nullptr, qb, NTOK, DM, DM);
    gemm_bf16<false, true, false><<<gemmDM, 256, gemmLds, stream>>>(lnb, wkt, bkp, nullptr, kb, NTOK, DM, DM);
    gemm_bf16<false, true, false><<<gemmDM, 256, gemmLds, stream>>>(lnb, wvt, bvp, nullptr, vb, NTOK, DM, DM);

    attn_flash<<<NB * NH * (SEQL / 64), 128, 0, stream>>>(qb, kb, vb, lnb);

    gemm_bf16<false, false, true><<<gemmDM, 256, gemmLds, stream>>>(lnb, wot, bo, xw, xw, NTOK, DM, DM);

    layernorm_k<true><<<NTOK, 256, 0, stream>>>(xw, n2g, n2b, lnb);

    gemm_bf16<true, true, false><<<gemmFF, 256, gemmLds, stream>>>(lnb, w1t, b1, nullptr, hb, NTOK, DFF, DM);

    gemm_bf16<false, false, true><<<gemmDM, 256, gemmLds, stream>>>(hb, w2t, b2, xw, xw, NTOK, DM, DFF);
  }

  layernorm_k<false><<<NTOK, 256, 0, stream>>>(
      xw, (const float*)d_in[34], (const float*)d_in[33], d_out);
}
